// Simple_34900904247875
// MI455X (gfx1250) — matrix operand order repaired and checked by reading, compiled, not run on hardware
//
#include <hip/hip_runtime.h>
#include <hip/hip_bf16.h>
#include <math.h>

typedef __attribute__((ext_vector_type(16))) _Float16 v16h;
typedef __attribute__((ext_vector_type(2)))  _Float16 h2;
typedef __attribute__((ext_vector_type(8)))  float    v8f;

#define ACT_BIAS   0
#define ACT_RELU   1
#define ACT_BNGELU 2

#if __has_builtin(__builtin_amdgcn_global_load_async_to_lds_b32)
#define ASYNC_LDS 1
#else
#define ASYNC_LDS 0
#endif

// async copy 4 bytes global -> LDS (no VGPR result, tracked by ASYNCcnt)
__device__ inline void async_b32(const void* g, void* l) {
#if ASYNC_LDS
  __builtin_amdgcn_global_load_async_to_lds_b32((int*)g, (int*)l, 0, 0);
#else
  *(int*)l = *(const int*)g;
#endif
}

// wait until <= N async-to-LDS ops outstanding (in-order retirement)
template <int N>
__device__ inline void wait_async() {
#if ASYNC_LDS
#if __has_builtin(__builtin_amdgcn_s_wait_asynccnt)
  __builtin_amdgcn_s_wait_asynccnt(N);
#else
  asm volatile("s_wait_asynccnt %0" ::"i"(N) : "memory");
#endif
#endif
}

// ---------------------------------------------------------------------------
// f32 -> f16 weight conversion
// ---------------------------------------------------------------------------
__global__ void cvt_f16_k(const float* __restrict__ s, _Float16* __restrict__ d, int n) {
  int i = blockIdx.x * blockDim.x + threadIdx.x;
  if (i < n) d[i] = (_Float16)s[i];
}

// ---------------------------------------------------------------------------
// 3x3 implicit-GEMM conv, Cout=64, pad=1, via V_WMMA_F32_16X16X32_F16.
// src is ZERO-PADDED [N][Cin][H+2][W+2] (origin (1,1)): no bounds checks in
// the hot loop. K = Cin*9 is a multiple of 32.
// Block = 256 threads = 8 waves; macro-tile 64 outch x 64 pixels.
// DOUBLE-BUFFERED async staging: each K-step issues the next step's
// 12 async b32 copies into the alternate LDS buffer, then
// s_wait_asynccnt(12) -- in-order retirement guarantees only the
// previous batch (current buffer) is forced complete, hiding latency.
// ---------------------------------------------------------------------------
__global__ __launch_bounds__(256) void conv3_wmma(
    const float* __restrict__ src, const _Float16* __restrict__ w16,
    const float* __restrict__ bias, const float* __restrict__ bn_g,
    const float* __restrict__ bn_b, float* __restrict__ dst,
    int Cin, int H, int W, int act, float bnmul,
    int dWp, int dOff, int dCh) {
  const int K   = Cin * 9;
  const int Wp  = W + 2;
  const int HWp = (H + 2) * Wp;
  const int HW  = H * W;
  const int tilesPerImg = HW >> 6;
  const int n  = blockIdx.x / tilesPerImg;
  const int p0 = (blockIdx.x % tilesPerImg) << 6;

  __shared__ float    Bsh[2][4][32][16];  // 16 KB ping-pong
  __shared__ _Float16 Ash[2][64][32];     //  8 KB ping-pong

  const int t  = threadIdx.x;
  const int wv = t >> 5;
  const int l  = t & 31;
  const int lm = l & 15;
  const int hi = l >> 4;

  const int og  = wv & 3;
  const int pgA = (wv >> 2) * 2;
  const int pgB = pgA + 1;

  const float* srcn = src + (size_t)n * Cin * HWp;

  // ---- hoisted staging slot coordinates ----
  // B: 2048 elements (4 tiles x 32 K x 16 px), 8 per thread.
  int   bK[8];
  const float* bBase[8];
  float* bLds[8];
#pragma unroll
  for (int i = 0; i < 8; ++i) {
    int e    = t + i * 256;
    int tile = e >> 9;
    int k    = (e >> 4) & 31;
    int p    = e & 15;
    int pix  = p0 + tile * 16 + p;
    int py   = pix / W, px = pix - py * W;
    bK[i]    = k;
    bBase[i] = srcn + (size_t)py * Wp + px;  // padded origin folded into r/3, r%3
    bLds[i]  = &Bsh[0][tile][k][p];
  }
  // A: 1024 h2 pairs (64 rows x 16 pairs), 4 per thread.
  const _Float16* aG[4];
  _Float16* aLds[4];
#pragma unroll
  for (int i = 0; i < 4; ++i) {
    int idx = t + i * 256;
    int m  = idx >> 4;
    int kp = (idx & 15) * 2;
    aG[i]   = w16 + (size_t)m * K + kp;
    aLds[i] = &Ash[0][m][kp];
  }

  // stage one K-step (12 async b32 per thread) into buffer `buf`
  auto stage = [&](int kk, int buf) {
#pragma unroll
    for (int i = 0; i < 8; ++i) {
      int kg = kk + bK[i];
      int c  = kg / 9;
      int r  = kg - c * 9;
      const float* g = bBase[i] + (size_t)c * HWp + (r / 3) * Wp + (r - (r / 3) * 3);
      async_b32(g, bLds[i] + buf * (4 * 32 * 16));
    }
#pragma unroll
    for (int i = 0; i < 4; ++i)
      async_b32(aG[i] + kk, aLds[i] + buf * (64 * 32));
  };

  // fragment gather patterns (ISA 7.12.2 16-bit layouts)
  const int am = 16 * og + lm;
  const int ak = hi ? 8 : 0;
  const int bk = hi ? 8 : 0;

  v8f accA = {}, accB = {};

  stage(0, 0);
  int buf = 0;
  for (int kk = 0; kk < K; kk += 32) {
    if (kk + 32 < K) {
      stage(kk + 32, buf ^ 1);  // prefetch next step into alternate buffer
      wait_async<12>();         // forces only the current buffer's batch
    } else {
      wait_async<0>();
    }
    __syncthreads();

    // ---- A fragment from LDS (2x ds_load_b128) ----
    v16h a;
    {
      const h2* arow = (const h2*)&Ash[buf][am][0];
#pragma unroll
      for (int r = 0; r < 4; ++r) {
        h2 v0 = arow[(ak + 2 * r) >> 1];
        h2 v1 = arow[(16 + ak + 2 * r) >> 1];
        a[2 * r + 0] = v0.x;
        a[2 * r + 1] = v0.y;
        a[2 * r + 8] = v1.x;
        a[2 * r + 9] = v1.y;
      }
    }
    // ---- B fragments from LDS + packed cvt ----
    v16h bA, bB;
#pragma unroll
    for (int j = 0; j < 16; ++j) {
      bA[j] = (_Float16)Bsh[buf][pgA][j < 8 ? bk + j : bk + 8 + j][lm];
      bB[j] = (_Float16)Bsh[buf][pgB][j < 8 ? bk + j : bk + 8 + j][lm];
    }

    accA = __builtin_amdgcn_wmma_f32_16x16x32_f16(false, a, false, bA,
                                                  (short)0, accA, false, false);
    accB = __builtin_amdgcn_wmma_f32_16x16x32_f16(false, a, false, bB,
                                                  (short)0, accB, false, false);
    __syncthreads();  // WAR: buffer staged next iter was gathered this iter
    buf ^= 1;
  }

  // ---- epilogue (D layout: VGPR r -> M = r + 8*hi, N = lane&15) ----
  const int mbase = 16 * og + 8 * hi;
#pragma unroll
  for (int tile = 0; tile < 2; ++tile) {
    v8f acc = tile ? accB : accA;
    int pix = p0 + (tile ? pgB : pgA) * 16 + lm;
    int py = pix / W, px = pix - py * W;
    float* dstn = dst + (size_t)n * 64 * dCh + dOff + (size_t)py * dWp + px;
#pragma unroll
    for (int r = 0; r < 8; ++r) {
      int m = mbase + r;
      float v = acc[r];
      if (act == ACT_BNGELU) {
        v = v * (bn_g[m] * bnmul) + bn_b[m];
        v = 0.5f * v * (1.0f + erff(v * 0.70710678118f));
      } else {
        v += bias[m];
        if (act == ACT_RELU) v = fmaxf(v, 0.0f);
      }
      dstn[(size_t)m * dCh] = v;
    }
  }
}

// ---------------------------------------------------------------------------
// 1x1 lateral conv (small), direct-gather WMMA.
// ---------------------------------------------------------------------------
__global__ __launch_bounds__(256) void conv1_wmma(
    const float* __restrict__ src, const _Float16* __restrict__ w16,
    const float* __restrict__ bias, float* __restrict__ dst,
    int Cin, int HW) {
  const int K = Cin;
  const int tilesPerImg = HW >> 6;
  const int n  = blockIdx.x / tilesPerImg;
  const int p0 = (blockIdx.x % tilesPerImg) << 6;

  const int t  = threadIdx.x;
  const int wv = t >> 5;
  const int l  = t & 31;
  const int lm = l & 15;
  const int hi = l >> 4;
  const int og  = wv & 3;
  const int pgA = (wv >> 2) * 2;
  const int pgB = pgA + 1;

  const float* srcn = src + (size_t)n * Cin * HW;
  const int pixA = p0 + pgA * 16 + lm;
  const int pixB = p0 + pgB * 16 + lm;

  const int am = 16 * og + lm;
  const int ak = hi ? 8 : 0;
  const h2* __restrict__ wrow = (const h2*)(w16 + (size_t)am * K);
  const int bk = hi ? 8 : 0;

  v8f accA = {}, accB = {};
  for (int kk = 0; kk < K; kk += 32) {
    v16h a;
#pragma unroll
    for (int r = 0; r < 8; ++r) {
      int kidx = (r < 4) ? (ak + 2 * r) : (16 + ak + 2 * (r - 4));
      int kg = kk + kidx;
      int ki = kg < K ? kg : K - 2;
      h2 v = wrow[ki >> 1];
      h2 z = {};
      v = (kg < K) ? v : z;
      a[2 * r + 0] = v.x;
      a[2 * r + 1] = v.y;
    }
    v16h bA, bB;
#pragma unroll
    for (int j = 0; j < 16; ++j) {
      int c = kk + (j < 8 ? bk + j : bk + 8 + j);
      int cc = c < K ? c : K - 1;
      float vA = srcn[(size_t)cc * HW + pixA];
      float vB = srcn[(size_t)cc * HW + pixB];
      bA[j] = (_Float16)(c < K ? vA : 0.0f);
      bB[j] = (_Float16)(c < K ? vB : 0.0f);
    }
    accA = __builtin_amdgcn_wmma_f32_16x16x32_f16(false, a, false, bA,
                                                  (short)0, accA, false, false);
    accB = __builtin_amdgcn_wmma_f32_16x16x32_f16(false, a, false, bB,
                                                  (short)0, accB, false, false);
  }
  const int mbase = 16 * og + 8 * hi;
#pragma unroll
  for (int tile = 0; tile < 2; ++tile) {
    v8f acc = tile ? accB : accA;
    float* dstn = dst + (size_t)n * 64 * HW + (tile ? pixB : pixA);
#pragma unroll
    for (int r = 0; r < 8; ++r)
      dstn[(size_t)(mbase + r) * HW] = acc[r] + bias[mbase + r];
  }
}

// ---------------------------------------------------------------------------
// Bilinear helper (pos = o*(n_in-1)/(n_out-1))
// ---------------------------------------------------------------------------
__device__ inline float bilerp(const float* __restrict__ s, int hs, int ws_,
                               int y, int x, int hd, int wd) {
  float fy = (hd > 1) ? (float)y * (float)(hs - 1) / (float)(hd - 1) : 0.0f;
  float fx = (wd > 1) ? (float)x * (float)(ws_ - 1) / (float)(wd - 1) : 0.0f;
  int ly = (int)floorf(fy);
  if (ly > hs - 2) ly = hs - 2;
  if (ly < 0) ly = 0;
  int lx = (int)floorf(fx);
  if (lx > ws_ - 2) lx = ws_ - 2;
  if (lx < 0) lx = 0;
  float ty = fy - (float)ly, tx = fx - (float)lx;
  const float* p = s + (size_t)ly * ws_ + lx;
  float v00 = p[0], v01 = p[1], v10 = p[ws_], v11 = p[ws_ + 1];
  return (1.0f - ty) * ((1.0f - tx) * v00 + tx * v01) +
         ty * ((1.0f - tx) * v10 + tx * v11);
}

// dst (PADDED, interior) = bilerp(src [N*64][hs][hs] -> hd x hd) + base (unpadded)
__global__ void upsample_add_k(const float* __restrict__ src,
                               const float* __restrict__ base,
                               float* __restrict__ dst, int N, int hs, int hd) {
  int idx = blockIdx.x * blockDim.x + threadIdx.x;
  int total = N * 64 * hd * hd;
  if (idx >= total) return;
  int x = idx % hd;
  int y = (idx / hd) % hd;
  int nc = idx / (hd * hd);
  const float* sp = src + (size_t)nc * hs * hs;
  int wp = hd + 2;
  dst[((size_t)nc * (hd + 2) + (y + 1)) * wp + (x + 1)] =
      bilerp(sp, hs, hs, y, x, hd, hd) + base[idx];
}

// concat [SM128, up(SM64), up(SM32), up(SM16), up(FS4)] -> PADDED 320ch @130x130
__global__ void upcat_k(const float* __restrict__ p128,
                        const float* __restrict__ p64,
                        const float* __restrict__ p32,
                        const float* __restrict__ p16,
                        const float* __restrict__ f8,
                        float* __restrict__ cat, int N) {
  int idx = blockIdx.x * blockDim.x + threadIdx.x;
  int total = N * 320 * 128 * 128;
  if (idx >= total) return;
  int x = idx & 127;
  int y = (idx >> 7) & 127;
  int cg = (idx >> 14) % 320;
  int n = idx / (320 << 14);
  int g = cg / 64, c = cg - g * 64;
  const float* srcs[5] = {p128, p64, p32, p16, f8};
  int hs = 128 >> g;
  const float* sp = srcs[g] + ((size_t)(n * 64 + c)) * hs * hs;
  float v = (g == 0) ? sp[(size_t)y * 128 + x] : bilerp(sp, hs, hs, y, x, 128, 128);
  cat[((size_t)(n * 320 + cg) * 130 + (y + 1)) * 130 + (x + 1)] = v;
}

// ROI max-pool 7x7 bins -> q[n][c] = sum_m mean_bins(max)   (z unpadded)
__global__ void roi_q_k(const float* __restrict__ z, const int* __restrict__ boxes,
                        float* __restrict__ q, int M) {
  int nm = blockIdx.x;
  int n = nm / M;
  const int* bx = boxes + (size_t)nm * 4;
  int xmin = bx[0], ymin = bx[1], rh = bx[2] + 1, rw = bx[3] + 1;
  const float* zn = z + (size_t)n * 64 * 128 * 128;
  for (int w = threadIdx.x; w < 64 * 49; w += blockDim.x) {
    int c = w / 49;
    int b = w - c * 49;
    int oy = b / 7, ox = b - oy * 7;
    int sh = oy * rh / 7, eh = ((oy + 1) * rh + 6) / 7;
    int sw = ox * rw / 7, ew = ((ox + 1) * rw + 6) / 7;
    const float* zc = zn + (size_t)c * 128 * 128;
    float mx = -3.4e38f;
    for (int ky = 0; ky < 10; ++ky) {
      if (sh + ky >= eh) break;
      int yy = ymin + sh + ky;
      yy = yy < 0 ? 0 : (yy > 127 ? 127 : yy);
      for (int kx = 0; kx < 10; ++kx) {
        if (sw + kx >= ew) break;
        int xx = xmin + sw + kx;
        xx = xx < 0 ? 0 : (xx > 127 ? 127 : xx);
        mx = fmaxf(mx, zc[(size_t)yy * 128 + xx]);
      }
    }
    atomicAdd(&q[n * 64 + c], mx * (1.0f / 49.0f));
  }
}

// att = sigmoid(sum_c q*z) (z unpadded); zatt written PADDED
__global__ void att_k(const float* __restrict__ z, const float* __restrict__ q,
                      float* __restrict__ att, float* __restrict__ zatt, int N) {
  int idx = blockIdx.x * blockDim.x + threadIdx.x;
  if (idx >= N * 16384) return;
  int n = idx >> 14;
  int yx = idx & 16383;
  int y = yx >> 7, x = yx & 127;
  const float* zn = z + (((size_t)n) << 14) * 64;
  const float* qn = q + n * 64;
  float s = 0.0f;
#pragma unroll 4
  for (int c = 0; c < 64; ++c) s += qn[c] * zn[(((size_t)c) << 14) + yx];
  float a = 1.0f / (1.0f + expf(-s));
  att[idx] = a;
  float* zo = zatt + (size_t)n * 64 * 130 * 130;
#pragma unroll 4
  for (int c = 0; c < 64; ++c)
    zo[((size_t)c * 130 + (y + 1)) * 130 + (x + 1)] =
        zn[(((size_t)c) << 14) + yx] * a;
}

// out1 = relu(1x1 conv 64->1)   (h unpadded)
__global__ void out1_k(const float* __restrict__ h, const float* __restrict__ w,
                       const float* __restrict__ b, float* __restrict__ out, int N) {
  int idx = blockIdx.x * blockDim.x + threadIdx.x;
  if (idx >= N * 16384) return;
  int n = idx >> 14;
  int yx = idx & 16383;
  const float* hn = h + (((size_t)n) << 14) * 64;
  float s = b[0];
#pragma unroll 4
  for (int c = 0; c < 64; ++c) s += w[c] * hn[(((size_t)c) << 14) + yx];
  out[idx] = fmaxf(s, 0.0f);
}

// offset = (2*sigmoid(1x1 conv 64->2)-1)*8   (g unpadded)
__global__ void offset_k(const float* __restrict__ g, const float* __restrict__ w,
                         const float* __restrict__ b, float* __restrict__ out, int N) {
  int idx = blockIdx.x * blockDim.x + threadIdx.x;
  if (idx >= N * 16384) return;
  int n = idx >> 14;
  int yx = idx & 16383;
  const float* gn = g + (((size_t)n) << 14) * 64;
#pragma unroll
  for (int oc = 0; oc < 2; ++oc) {
    float s = b[oc];
#pragma unroll 4
    for (int c = 0; c < 64; ++c) s += w[oc * 64 + c] * gn[(((size_t)c) << 14) + yx];
    float sg = 1.0f / (1.0f + expf(-s));
    out[((((size_t)n) * 2 + oc) << 14) + yx] = (2.0f * sg - 1.0f) * 8.0f;
  }
}

// ---------------------------------------------------------------------------
// Host orchestration
// ---------------------------------------------------------------------------
extern "C" void kernel_launch(void* const* d_in, const int* in_sizes, int n_in,
                              void* d_out, int out_size, void* d_ws, size_t ws_size,
                              hipStream_t stream) {
  (void)n_in; (void)out_size; (void)ws_size;
  const int N = in_sizes[0] / (64 * 128 * 128);  // = 4

  const float* x0 = (const float*)d_in[0];
  const float* x1 = (const float*)d_in[1];
  const float* x2 = (const float*)d_in[2];
  const float* x3 = (const float*)d_in[3];
  const float* x4 = (const float*)d_in[4];
  const int*   boxes = (const int*)d_in[5];
  // dict leaves flattened in sorted key order (JAX pytree convention).
  const int FB1 = 6, FB2 = 19, HB = 32;

  // ---- workspace carve-up (floats) ----
  float* Wp = (float*)d_ws;
  size_t cur = 0;
  auto alloc = [&](size_t nf) { float* p = Wp + cur; cur += nf; return p; };
  auto padsz = [&](int C, int H) { return (size_t)N * C * (H + 2) * (H + 2); };
  // unpadded f32 buffers
  float* FS1 = alloc((size_t)N * 64 * 64 * 64);
  float* FS2 = alloc((size_t)N * 64 * 32 * 32);
  float* FS3 = alloc((size_t)N * 64 * 16 * 16);
  float* FS4 = alloc((size_t)N * 64 * 8 * 8);
  float* SM128 = alloc((size_t)N * 64 * 128 * 128);
  float* SM64  = alloc((size_t)N * 64 * 64 * 64);
  float* SM32  = alloc((size_t)N * 64 * 32 * 32);
  float* SM16  = alloc((size_t)N * 64 * 16 * 16);
  float* Z1  = alloc((size_t)N * 64 * 128 * 128);
  float* HB_ = alloc((size_t)N * 64 * 128 * 128);
  float* Q   = alloc(256);
  // PADDED f32 buffers (contiguous: one memset zeroes all halos)
  float* PADBASE = Wp + cur;
  float* PRE128 = alloc(padsz(64, 128));
  float* PRE64  = alloc(padsz(64, 64));
  float* PRE32  = alloc(padsz(64, 32));
  float* PRE16  = alloc(padsz(64, 16));
  float* CAT  = alloc(padsz(320, 128));
  float* ZATT = alloc(padsz(64, 128));
  float* HA   = alloc(padsz(64, 128));
  float* Z2   = alloc(padsz(64, 128));
  size_t padTotal = (size_t)(Wp + cur - PADBASE);
  // f16 weights
  _Float16* w16 = (_Float16*)alloc(210000);
  _Float16* W_c10 = w16;
  _Float16* W_c11 = W_c10 + 64 * 48;
  _Float16* W_c12 = W_c11 + 64 * 96;
  _Float16* W_c13 = W_c12 + 64 * 192;
  _Float16* W_sm  = W_c13 + 64 * 384;
  _Float16* W_fu  = W_sm + 64 * 64 * 9;
  _Float16* W_lw1 = W_fu + 64 * 320 * 9;
  _Float16* W_lw2 = W_lw1 + 64 * 64 * 9;
  _Float16* W_ow1 = W_lw2 + 64 * 64 * 9;
  _Float16* W_ow2 = W_ow1 + 64 * 64 * 9;

  // zero all padded buffers (halos) once per call
  (void)hipMemsetAsync(PADBASE, 0, padTotal * sizeof(float), stream);
  (void)hipMemsetAsync(Q, 0, 256 * sizeof(float), stream);

  auto cvt = [&](const void* s, _Float16* d, int n) {
    cvt_f16_k<<<(n + 255) / 256, 256, 0, stream>>>((const float*)s, d, n);
  };
  auto conv1 = [&](const float* src, const _Float16* w, const void* bias,
                   float* dst, int Cin, int H) {
    conv1_wmma<<<N * H * H / 64, 256, 0, stream>>>(
        src, w, (const float*)bias, dst, Cin, H * H);
  };
  // src padded; dst padded or not
  auto conv3 = [&](const float* src, const _Float16* w, const void* bias,
                   float* dst, int Cin, int H, int act, const void* g,
                   const void* b, float bnmul, bool dstPadded) {
    int dWp = dstPadded ? H + 2 : H;
    int dOff = dstPadded ? dWp + 1 : 0;
    int dCh = dstPadded ? (H + 2) * dWp : H * H;
    conv3_wmma<<<N * H * H / 64, 256, 0, stream>>>(
        src, w, (const float*)bias, (const float*)g, (const float*)b, dst,
        Cin, H, H, act, bnmul, dWp, dOff, dCh);
  };

  const float BNMUL = 0.99999500003750f;  // 1/sqrt(1 + 1e-5)

  // ---- convert head weights once ----
  cvt(d_in[HB + 3], W_lw1, 64 * 64 * 9);
  cvt(d_in[HB + 4], W_lw2, 64 * 64 * 9);
  cvt(d_in[HB + 9], W_ow1, 64 * 64 * 9);
  cvt(d_in[HB + 10], W_ow2, 64 * 64 * 9);

  auto run_fuse = [&](int fb, float* zout, bool zPadded) {
    cvt(d_in[fb + 3], W_c10, 64 * 48);
    cvt(d_in[fb + 5], W_c11, 64 * 96);
    cvt(d_in[fb + 7], W_c12, 64 * 192);
    cvt(d_in[fb + 9], W_c13, 64 * 384);
    cvt(d_in[fb + 12], W_sm, 64 * 64 * 9);
    cvt(d_in[fb + 10], W_fu, 64 * 320 * 9);
    // lateral 1x1 convs -> unpadded FS
    conv1(x1, W_c10, d_in[fb + 2], FS1, 48, 64);
    conv1(x2, W_c11, d_in[fb + 4], FS2, 96, 32);
    conv1(x3, W_c12, d_in[fb + 6], FS3, 192, 16);
    conv1(x4, W_c13, d_in[fb + 8], FS4, 384, 8);
    // upsample + add -> PADDED PRE
    auto up = [&](const float* s, const float* base, float* d, int hs, int hd) {
      int tot = N * 64 * hd * hd;
      upsample_add_k<<<(tot + 255) / 256, 256, 0, stream>>>(s, base, d, N, hs, hd);
    };
    up(FS4, FS3, PRE16, 8, 16);
    up(FS3, FS2, PRE32, 16, 32);
    up(FS2, FS1, PRE64, 32, 64);
    up(FS1, x0, PRE128, 64, 128);
    // smooth 3x3 convs (padded -> unpadded SM)
    const void* smb = d_in[fb + 11];
    conv3(PRE16, W_sm, smb, SM16, 64, 16, ACT_BIAS, nullptr, nullptr, 1.0f, false);
    conv3(PRE32, W_sm, smb, SM32, 64, 32, ACT_BIAS, nullptr, nullptr, 1.0f, false);
    conv3(PRE64, W_sm, smb, SM64, 64, 64, ACT_BIAS, nullptr, nullptr, 1.0f, false);
    conv3(PRE128, W_sm, smb, SM128, 64, 128, ACT_BIAS, nullptr, nullptr, 1.0f, false);
    // upsample-to-128 + concat -> PADDED CAT (320ch)
    {
      int tot = N * 320 * 128 * 128;
      upcat_k<<<(tot + 255) / 256, 256, 0, stream>>>(SM128, SM64, SM32, SM16, FS4, CAT, N);
    }
    // fuse conv 3x3 320->64 + BN-scale + GELU
    conv3(CAT, W_fu, nullptr, zout, 320, 128, ACT_BNGELU,
          d_in[fb + 1], d_in[fb + 0], BNMUL, zPadded);
  };

  float* out1_p   = (float*)d_out;
  float* offset_p = (float*)d_out + (size_t)N * 16384;
  float* att_p    = (float*)d_out + (size_t)N * 3 * 16384;

  // ---- pass 1: z1 (unpadded), ROI attention, l-head ----
  run_fuse(FB1, Z1, false);
  roi_q_k<<<N * 32, 256, 0, stream>>>(Z1, boxes, Q, 32);
  {
    int tot = N * 16384;
    att_k<<<(tot + 255) / 256, 256, 0, stream>>>(Z1, Q, att_p, ZATT, N);
  }
  conv3(ZATT, W_lw1, d_in[HB + 0], HA, 64, 128, ACT_RELU, nullptr, nullptr, 1.0f, true);
  conv3(HA, W_lw2, d_in[HB + 1], HB_, 64, 128, ACT_RELU, nullptr, nullptr, 1.0f, false);
  {
    int tot = N * 16384;
    out1_k<<<(tot + 255) / 256, 256, 0, stream>>>(
        HB_, (const float*)d_in[HB + 5], (const float*)d_in[HB + 2], out1_p, N);
  }

  // ---- pass 2: z2 (padded), o-head ----
  run_fuse(FB2, Z2, true);
  conv3(Z2, W_ow1, d_in[HB + 6], HA, 64, 128, ACT_RELU, nullptr, nullptr, 1.0f, true);
  conv3(HA, W_ow2, d_in[HB + 7], HB_, 64, 128, ACT_RELU, nullptr, nullptr, 1.0f, false);
  {
    int tot = N * 16384;
    offset_k<<<(tot + 255) / 256, 256, 0, stream>>>(
        HB_, (const float*)d_in[HB + 11], (const float*)d_in[HB + 8], offset_p, N);
  }
}
